// AgentEncoder_43456479101524
// MI455X (gfx1250) — matrix operand order repaired and checked by reading, not compiled, not run on hardware
//
#include <hip/hip_runtime.h>
#include <hip/hip_bf16.h>
#include <math.h>

// ---------------- constants from the reference ----------------
#define NA_ 192
#define T_  21
#define NP_ 512
#define D_  128
#define H_  8
#define NL_ 3

typedef __bf16 bf16;
typedef __attribute__((ext_vector_type(16))) __bf16 v16bf;
typedef __attribute__((ext_vector_type(8)))  __bf16 bf16x8;
typedef __attribute__((ext_vector_type(4)))  __bf16 bf16x4;
typedef __attribute__((ext_vector_type(8)))  float  v8f;

// ==================== device helpers ====================

__device__ __forceinline__ float wrap_ang(float a) {
    const float PI_ = 3.14159265358979323846f;
    float r = fmodf(a + PI_, 2.0f * PI_);
    if (r < 0.0f) r += 2.0f * PI_;
    return r - PI_;
}
__device__ __forceinline__ float safe_angle(float x, float y) {
    bool degen = (fabsf(x) + fabsf(y)) < 1e-9f;
    return atan2f(degen ? 0.0f : y, degen ? 1.0f : x);
}
__device__ __forceinline__ float angle_between(float ux, float uy, float vx, float vy) {
    float cr = ux * vy - uy * vx;
    float dt = ux * vx + uy * vy;
    bool degen = (fabsf(cr) + fabsf(dt)) < 1e-9f;
    return atan2f(degen ? 0.0f : cr, degen ? 1.0f : dt);
}
__device__ __forceinline__ void atomicMaxF(float* addr, float val) {
    unsigned int* u = (unsigned int*)addr;
    unsigned int cur = *u;
    while (__uint_as_float(cur) < val) {
        unsigned int prev = atomicCAS(u, cur, __float_as_uint(val));
        if (prev == cur) break;
        cur = prev;
    }
}

// ==================== generic kernels ====================

__global__ void fill_f32(float* p, float v, long n) {
    long i = (long)blockIdx.x * blockDim.x + threadIdx.x;
    if (i < n) p[i] = v;
}

__global__ void cvt_f32_bf16(const float* x, bf16* y, long n) {
    long i = (long)blockIdx.x * blockDim.x + threadIdx.x;
    if (i < n) y[i] = (bf16)x[i];
}

// weight: fp32 row-major [K,N] -> bf16 transposed [N,Kp] (zero-padded K)
__global__ void convert_wT(const float* w, bf16* wt, int K, int N, int Kp) {
    long i = (long)blockIdx.x * blockDim.x + threadIdx.x;
    if (i >= (long)N * Kp) return;
    int n = (int)(i / Kp), k = (int)(i % Kp);
    wt[i] = (k < K) ? (bf16)w[(long)k * N + n] : (bf16)0.0f;
}

__global__ void concat_bias_k(const float* b1, const float* b2, float* o) {
    int i = blockIdx.x * blockDim.x + threadIdx.x;
    if (i < 256) o[i] = (i < 128) ? b1[i] : b2[i - 128];
}

// ==================== WMMA GEMM ====================
// C[M,N] = A[M,K](bf16) @ Bt[N,K](bf16, pre-transposed) + bias (+ residual) (opt relu)
// Block = 256 threads = 8 waves; each wave owns a 16(M) x 128(N) strip (8 acc).
// Weight chunks (32K x 128N, 8 KB) are staged through a DOUBLE-BUFFERED LDS
// pipeline using CDNA5 GLOBAL_LOAD_ASYNC_TO_LDS_B128 (ASYNCcnt), so HBM latency
// overlaps the 8-WMMA compute block. Padded stride 40 bf16 -> each half-wave's
// ds_load_b128 fragment reads hit all 64 LDS banks exactly once.
#define LDS_STRIDE 40

__device__ __forceinline__ void stage_async(const bf16* __restrict__ Bt, int nbase,
                                            int K, int k0, bf16* dst, int tid) {
    int row = tid >> 2, seg = tid & 3;        // rows 0..63 (+64), 4 x 16B segments
    {
        unsigned l0 = (unsigned)(size_t)(dst + row * LDS_STRIDE + seg * 8);
        const bf16* g0 = Bt + (long)(nbase + row) * K + k0 + seg * 8;
        asm volatile("global_load_async_to_lds_b128 %0, %1, off"
                     :: "v"(l0), "v"(g0) : "memory");
    }
    {
        unsigned l1 = (unsigned)(size_t)(dst + (row + 64) * LDS_STRIDE + seg * 8);
        const bf16* g1 = Bt + (long)(nbase + row + 64) * K + k0 + seg * 8;
        asm volatile("global_load_async_to_lds_b128 %0, %1, off"
                     :: "v"(l1), "v"(g1) : "memory");
    }
}

__global__ void __launch_bounds__(256) gemm_bf16_wmma(
        const bf16* __restrict__ A, const bf16* __restrict__ Bt,
        const float* __restrict__ bias, const float* residual,
        float* Cf, bf16* Cbf, int M, int N, int K, int relu) {
    __shared__ bf16 bsh[2][128 * LDS_STRIDE];   // 2 x 10 KB

    int tid   = threadIdx.x;
    int wave  = tid >> 5;
    int lane  = tid & 31;
    int mtile = blockIdx.y * 8 + wave;
    int nbase = blockIdx.x * 128;

    int r    = lane & 15;
    int half = lane >> 4;
    int kbA  = half * 8;     // A: elems 0..7 -> K=kbA..+7, elems 8..15 -> K=kbA+16..+23
    int kbB  = half * 8;     // B: elems 0..7 -> K=kbB..+7, elems 8..15 -> K=kbB+16..+23

    long arow = (long)mtile * 16 + r;
    bool mlive = (mtile * 16) < M;
    if (arow >= M) arow = M - 1;                 // clamp; stores are guarded
    const bf16* pa = A + arow * (long)K + kbA;

    v8f acc[8] = {};

    // prologue: stage first chunk
    stage_async(Bt, nbase, K, 0, &bsh[0][0], tid);
    asm volatile("s_wait_asynccnt 0x0" ::: "memory");
    __syncthreads();

    int buf = 0;
    for (int k0 = 0; k0 < K; k0 += 32) {
        int nxt = buf ^ 1;
        if (k0 + 32 < K) stage_async(Bt, nbase, K, k0 + 32, &bsh[nxt][0], tid);

        __builtin_prefetch(pa + k0 + 64, 0, 0);
        bf16x8 alo = *(const bf16x8*)(pa + k0);
        bf16x8 ahi = *(const bf16x8*)(pa + k0 + 16);
        v16bf av = __builtin_shufflevector(alo, ahi, 0, 1, 2, 3, 4, 5, 6, 7,
                                           8, 9, 10, 11, 12, 13, 14, 15);
#pragma unroll
        for (int j = 0; j < 8; ++j) {
            const bf16* pb = &bsh[buf][(j * 16 + r) * LDS_STRIDE + kbB];
            bf16x8 blo = *(const bf16x8*)(pb);
            bf16x8 bhi = *(const bf16x8*)(pb + 16);
            v16bf bv = __builtin_shufflevector(blo, bhi, 0, 1, 2, 3, 4, 5, 6, 7,
                                               8, 9, 10, 11, 12, 13, 14, 15);
            acc[j] = __builtin_amdgcn_wmma_f32_16x16x32_bf16(
                false, av, false, bv, (short)0, acc[j], false, false);
        }

        asm volatile("s_wait_asynccnt 0x0" ::: "memory");
        __syncthreads();          // next chunk staged; current fully consumed
        buf = nxt;
    }

    if (!mlive) return;
#pragma unroll
    for (int j = 0; j < 8; ++j) {
        int n = nbase + j * 16 + r;
        if (n >= N) continue;
        float bsum = bias ? bias[n] : 0.0f;
#pragma unroll
        for (int rr = 0; rr < 8; ++rr) {
            int m = mtile * 16 + rr + half * 8;   // C/D layout: VGPR rr, half selects +8
            if (m < M) {
                float v = acc[j][rr] + bsum;
                if (residual) v += residual[(long)m * N + n];
                if (relu) v = v > 0.0f ? v : 0.0f;
                if (Cf)  Cf[(long)m * N + n] = v;
                if (Cbf) Cbf[(long)m * N + n] = (bf16)v;
            }
        }
    }
}

// ==================== LayerNorm over D=128 (one wave per row) ====================
__global__ void layernorm128(const float* __restrict__ x, const float* __restrict__ g,
                             const float* __restrict__ bb,
                             float* yf, bf16* ybf, int M, int relu) {
    int row  = (int)(((long)blockIdx.x * blockDim.x + threadIdx.x) >> 5);
    int lane = threadIdx.x & 31;
    if (row >= M) return;
    float4 f = ((const float4*)(x + (long)row * 128))[lane];
    float s = f.x + f.y + f.z + f.w;
    for (int o = 16; o > 0; o >>= 1) s += __shfl_xor(s, o, 32);
    float mu = s * (1.0f / 128.0f);
    float dx = f.x - mu, dy = f.y - mu, dz = f.z - mu, dw = f.w - mu;
    float vs = dx * dx + dy * dy + dz * dz + dw * dw;
    for (int o = 16; o > 0; o >>= 1) vs += __shfl_xor(vs, o, 32);
    float inv = rsqrtf(vs * (1.0f / 128.0f) + 1e-5f);
    float4 gg = ((const float4*)g)[lane];
    float4 bv = ((const float4*)bb)[lane];
    float4 o4;
    o4.x = dx * inv * gg.x + bv.x;
    o4.y = dy * inv * gg.y + bv.y;
    o4.z = dz * inv * gg.z + bv.z;
    o4.w = dw * inv * gg.w + bv.w;
    if (relu) {
        o4.x = fmaxf(o4.x, 0.0f); o4.y = fmaxf(o4.y, 0.0f);
        o4.z = fmaxf(o4.z, 0.0f); o4.w = fmaxf(o4.w, 0.0f);
    }
    if (yf) ((float4*)(yf + (long)row * 128))[lane] = o4;
    if (ybf) {
        bf16x4 h;
        h[0] = (bf16)o4.x; h[1] = (bf16)o4.y; h[2] = (bf16)o4.z; h[3] = (bf16)o4.w;
        *(bf16x4*)(ybf + (long)row * 128 + lane * 4) = h;
    }
}

// ==================== feature kernels ====================

__global__ void state_feat_k(const float* pos, const float* head, const float* vel, bf16* pad) {
    int i = blockIdx.x * blockDim.x + threadIdx.x;
    if (i >= NA_ * T_) return;
    int t = i % T_;
    float h = head[i];
    float hx = cosf(h), hy = sinf(h);
    float mx = 0.0f, my = 0.0f;
    if (t > 0) {
        mx = pos[2 * i] - pos[2 * (i - 1)];
        my = pos[2 * i + 1] - pos[2 * (i - 1) + 1];
    }
    float vx = vel[2 * i], vy = vel[2 * i + 1];
    bf16* o = pad + (long)i * 32;
    o[0] = (bf16)sqrtf(mx * mx + my * my + 1e-12f);
    o[1] = (bf16)angle_between(hx, hy, mx, my);
    o[2] = (bf16)sqrtf(vx * vx + vy * vy + 1e-12f);
    o[3] = (bf16)angle_between(hx, hy, vx, vy);
}

__global__ void box_pad_k(const float* box, bf16* pad) {
    int i = blockIdx.x * blockDim.x + threadIdx.x;
    if (i >= NA_ * 4) return;
    pad[(long)(i / 4) * 32 + (i % 4)] = (bf16)box[i];
}

__global__ void add_static_k(float* emb, const float* stat, const float* temb,
                             const float* idemb, const int* atype, const int* aid) {
    long i = (long)blockIdx.x * blockDim.x + threadIdx.x;
    if (i >= (long)NA_ * T_ * 128) return;
    int c = (int)(i & 127);
    int node = (int)(i >> 7);
    int a = node / T_;
    emb[i] += stat[(long)a * 128 + c] + temb[(long)atype[a] * 128 + c] +
              idemb[(long)aid[a] * 128 + c];
}

__global__ void temporal_feat_k(const int* src, const int* dst, const float* pos,
                                const float* head, bf16* pad, int E) {
    int e = blockIdx.x * blockDim.x + threadIdx.x;
    if (e >= E) return;
    int s = src[e], d = dst[e];
    float hd = head[d], c = cosf(hd), sn = sinf(hd);
    float dx = pos[2 * s] - pos[2 * d], dy = pos[2 * s + 1] - pos[2 * d + 1];
    float rx = dx * c + dy * sn, ry = -dx * sn + dy * c;
    float l = sqrtf(rx * rx + ry * ry + 1e-12f);
    float th = safe_angle(rx, ry);
    float hdf = wrap_ang(head[s] - hd);
    bf16* o = pad + (long)e * 32;
    o[0] = (bf16)l; o[1] = (bf16)cosf(th); o[2] = (bf16)sinf(th);
    o[3] = (bf16)cosf(hdf); o[4] = (bf16)sinf(hdf);
    o[5] = (bf16)(float)((d % T_) - (s % T_));
}

__global__ void map_feat_k(const int* src, const int* dst, const float* ppos,
                           const float* phead, const float* hvalid,
                           const float* pos, const float* head, bf16* pad, int E) {
    int e = blockIdx.x * blockDim.x + threadIdx.x;
    if (e >= E) return;
    int s = src[e], d = dst[e];
    float hd = head[d], c = cosf(hd), sn = sinf(hd);
    float dx = ppos[2 * s] - pos[2 * d], dy = ppos[2 * s + 1] - pos[2 * d + 1];
    float rx = dx * c + dy * sn, ry = -dx * sn + dy * c;
    float l = sqrtf(rx * rx + ry * ry + 1e-12f);
    float th = safe_angle(rx, ry);
    float hdf = wrap_ang(phead[s] - hd);
    bf16* o = pad + (long)e * 32;
    o[0] = (bf16)l; o[1] = (bf16)cosf(th); o[2] = (bf16)sinf(th);
    o[3] = (bf16)cosf(hdf); o[4] = (bf16)sinf(hdf); o[5] = (bf16)hvalid[s];
}

__global__ void agent_feat_k(const int* src, const int* dst, const float* pos,
                             const float* head, bf16* pad, int E) {
    int e = blockIdx.x * blockDim.x + threadIdx.x;
    if (e >= E) return;
    int s = src[e], d = dst[e];
    int sf = (s % NA_) * T_ + (s / NA_);   // time-major -> agent-major flat
    int df = (d % NA_) * T_ + (d / NA_);
    float hd = head[df], c = cosf(hd), sn = sinf(hd);
    float dx = pos[2 * sf] - pos[2 * df], dy = pos[2 * sf + 1] - pos[2 * df + 1];
    float rx = dx * c + dy * sn, ry = -dx * sn + dy * c;
    float l = sqrtf(rx * rx + ry * ry + 1e-12f);
    float th = safe_angle(rx, ry);
    float hdf = wrap_ang(head[sf] - hd);
    bf16* o = pad + (long)e * 32;
    o[0] = (bf16)l; o[1] = (bf16)cosf(th); o[2] = (bf16)sinf(th);
    o[3] = (bf16)cosf(hdf); o[4] = (bf16)sinf(hdf);
}

// ==================== attention kernels ====================

__global__ void gather_kin_k(const bf16* xs, const bf16* ea, const int* src,
                             bf16* kin, int E) {
    long i = (long)blockIdx.x * blockDim.x + threadIdx.x;
    if (i >= (long)E * 256) return;
    int e = (int)(i >> 8), c = (int)(i & 255);
    kin[i] = (c < 128) ? xs[(long)src[e] * 128 + c] : ea[(long)e * 128 + (c - 128)];
}

// kv layout per edge: [0:128) = K, [128:256) = V
__global__ void edge_logits_k(const float* q, const float* kv, const int* dst,
                              float* lg, int E) {
    long i = (long)blockIdx.x * blockDim.x + threadIdx.x;
    if (i >= (long)E * 8) return;
    int e = (int)(i >> 3), h = (int)(i & 7);
    const float* qp = q + (long)dst[e] * 128 + h * 16;
    const float* kp = kv + (long)e * 256 + h * 16;
    float s = 0.0f;
#pragma unroll
    for (int j = 0; j < 16; ++j) s += qp[j] * kp[j];
    lg[i] = s * 0.25f;   // 1/sqrt(16)
}

__global__ void seg_max_k(const float* lg, const int* dst, float* m, int E) {
    long i = (long)blockIdx.x * blockDim.x + threadIdx.x;
    if (i >= (long)E * 8) return;
    int e = (int)(i >> 3), h = (int)(i & 7);
    atomicMaxF(&m[(long)dst[e] * 8 + h], lg[i]);
}

__global__ void seg_expsum_k(const float* lg, const int* dst, const float* m,
                             float* ex, float* den, int E) {
    long i = (long)blockIdx.x * blockDim.x + threadIdx.x;
    if (i >= (long)E * 8) return;
    int e = (int)(i >> 3), h = (int)(i & 7);
    float v = expf(lg[i] - m[(long)dst[e] * 8 + h]);
    ex[i] = v;
    atomicAdd(&den[(long)dst[e] * 8 + h], v);
}

__global__ void seg_scatter_k(const float* ex, const float* den, const float* kv,
                              const int* dst, float* agg, int E) {
    long i = (long)blockIdx.x * blockDim.x + threadIdx.x;
    if (i >= (long)E * 128) return;
    int e = (int)(i >> 7), c = (int)(i & 127), h = c >> 4;
    long d = dst[e];
    float w = ex[(long)e * 8 + h] / den[d * 8 + h];
    atomicAdd(&agg[d * 128 + c], w * kv[(long)e * 256 + 128 + c]);
}

__global__ void transpose_nodes_k(const float* x, float* y, int toTimeMajor) {
    long i = (long)blockIdx.x * blockDim.x + threadIdx.x;
    if (i >= (long)NA_ * T_ * 128) return;
    int c = (int)(i & 127);
    int node = (int)(i >> 7);
    if (toTimeMajor) {
        int a = node / T_, t = node % T_;
        y[((long)(t * NA_ + a) << 7) + c] = x[i];
    } else {
        int t = node / NA_, a = node % NA_;
        y[((long)(a * T_ + t) << 7) + c] = x[i];
    }
}

__global__ void write_out_k(const float* emb, const unsigned char* vis, float* out,
                            long n_emb, long n_mask, long out_n) {
    long i = (long)blockIdx.x * blockDim.x + threadIdx.x;
    if (i >= out_n) return;
    if (i < n_emb) out[i] = emb[i];
    else if (i < n_emb + n_mask) out[i] = vis[i - n_emb] ? 1.0f : 0.0f;
}

// ==================== host-side plumbing ====================

struct Bump {
    char* b; size_t o, cap;
    void* get(size_t n) { size_t a = (o + 255) & ~(size_t)255; o = a + n; return b + a; }
};

struct Cur { void* const* d; int i; const float* nf() { return (const float*)d[i++]; } };
struct LinP { const float *w, *b; };
struct LnP  { const float *g, *b; };
struct MlpP { LinP l1, l2; LnP ln; };
struct AttnPRaw { LinP f1, f2, k, o, q, v; LnP ln_dst, ln_edge, ln_ff, ln_src; };

// JAX pytree flatten: dict keys in sorted order.
static LinP parse_lin(Cur& c) { LinP p; p.b = c.nf(); p.w = c.nf(); return p; }       // b,w
static LnP  parse_ln (Cur& c) { LnP p; p.b = c.nf(); p.g = c.nf(); return p; }        // b,g
static MlpP parse_mlp(Cur& c) { MlpP m; m.l1 = parse_lin(c); m.l2 = parse_lin(c); m.ln = parse_ln(c); return m; }  // l1,l2,ln
static AttnPRaw parse_attn(Cur& c) {                                                  // f1,f2,k,ln_dst,ln_edge,ln_ff,ln_src,o,q,v
    AttnPRaw a;
    a.f1 = parse_lin(c); a.f2 = parse_lin(c); a.k = parse_lin(c);
    a.ln_dst = parse_ln(c); a.ln_edge = parse_ln(c); a.ln_ff = parse_ln(c); a.ln_src = parse_ln(c);
    a.o = parse_lin(c); a.q = parse_lin(c); a.v = parse_lin(c);
    return a;
}

static inline int cdiv(long a, long b) { return (int)((a + b - 1) / b); }

struct LinW { bf16* wT; const float* b; int Kp; };
struct MlpW { LinW l1, l2; LnP ln; };
struct AttnW {
    LinW q, o, f1, f2;
    bf16* kvT;            // [256 out rows][256 K] = [Wk;Wv] transposed
    float* kvb;           // concat bias (device, 256)
    LnP ln_src, ln_dst, ln_edge, ln_ff;
};

static LinW conv_lin(Bump& ws, hipStream_t s, LinP p, int K, int N, int Kp) {
    LinW w; w.b = p.b; w.Kp = Kp;
    w.wT = (bf16*)ws.get((size_t)N * Kp * sizeof(bf16));
    long tot = (long)N * Kp;
    convert_wT<<<cdiv(tot, 256), 256, 0, s>>>(p.w, w.wT, K, N, Kp);
    return w;
}
static MlpW conv_mlp(Bump& ws, hipStream_t s, const MlpP& m, int din) {
    MlpW w; w.l1 = conv_lin(ws, s, m.l1, din, 128, 32);
    w.l2 = conv_lin(ws, s, m.l2, 128, 128, 128); w.ln = m.ln; return w;
}
static AttnW conv_attn(Bump& ws, hipStream_t s, const AttnPRaw& a) {
    AttnW w;
    w.q  = conv_lin(ws, s, a.q, 128, 128, 128);
    w.o  = conv_lin(ws, s, a.o, 128, 128, 128);
    w.f1 = conv_lin(ws, s, a.f1, 128, 512, 128);
    w.f2 = conv_lin(ws, s, a.f2, 512, 128, 512);
    // fused KV: rows 0..127 = Wk^T, rows 128..255 = Wv^T  (K = 256)
    w.kvT = (bf16*)ws.get((size_t)256 * 256 * sizeof(bf16));
    convert_wT<<<cdiv((long)128 * 256, 256), 256, 0, s>>>(a.k.w, w.kvT, 256, 128, 256);
    convert_wT<<<cdiv((long)128 * 256, 256), 256, 0, s>>>(a.v.w, w.kvT + 128 * 256, 256, 128, 256);
    w.kvb = (float*)ws.get(256 * sizeof(float));
    concat_bias_k<<<1, 256, 0, s>>>(a.k.b, a.v.b, w.kvb);
    w.ln_src = a.ln_src; w.ln_dst = a.ln_dst; w.ln_edge = a.ln_edge; w.ln_ff = a.ln_ff;
    return w;
}

static void gemm(hipStream_t s, const bf16* A, const bf16* Bt, const float* bias,
                 const float* res, float* Cf, bf16* Cbf, int M, int N, int K, int relu) {
    if (M <= 0) return;
    dim3 g(N / 128, cdiv(M, 128));                     // N is 128/256/512 here
    gemm_bf16_wmma<<<g, 256, 0, s>>>(A, Bt, bias, res, Cf, Cbf, M, N, K, relu);
}
static void lnorm(hipStream_t s, const float* x, const LnP& p, float* yf, bf16* ybf,
                  int M, int relu) {
    if (M <= 0) return;
    layernorm128<<<cdiv(M, 8), 256, 0, s>>>(x, p.g, p.b, yf, ybf, M, relu);
}
static void fill(hipStream_t s, float* p, float v, long n) {
    if (n > 0) fill_f32<<<cdiv(n, 256), 256, 0, s>>>(p, v, n);
}

static void run_mlp2(hipStream_t s, const MlpW& w, const bf16* inPad, float* out, int M,
                     float* tf, bf16* tb) {
    gemm(s, inPad, w.l1.wT, w.l1.b, nullptr, tf, nullptr, M, 128, 32, 0);
    lnorm(s, tf, w.ln, nullptr, tb, M, 1);               // relu(LayerNorm(.))
    gemm(s, tb, w.l2.wT, w.l2.b, nullptr, out, nullptr, M, 128, 128, 0);
}

struct Scr {
    bf16 *xs_bf, *xd_bf, *ea_bf, *kin, *agg_bf, *h1_bf;
    float *qf, *kvf, *logits, *ex, *m, *den, *agg;
};

static void run_attention(hipStream_t s, const AttnW& w,
                          const float* xsrc, int Ns, float* x, int Nd,
                          const int* esrc, const int* edst, int E,
                          const float* eattr, const Scr& sc) {
    if (E <= 0) return;
    lnorm(s, xsrc, w.ln_src, nullptr, sc.xs_bf, Ns, 0);
    lnorm(s, x,    w.ln_dst, nullptr, sc.xd_bf, Nd, 0);
    lnorm(s, eattr, w.ln_edge, nullptr, sc.ea_bf, E, 0);
    gemm(s, sc.xd_bf, w.q.wT, w.q.b, nullptr, sc.qf, nullptr, Nd, 128, 128, 0);
    gather_kin_k<<<cdiv((long)E * 256, 256), 256, 0, s>>>(sc.xs_bf, sc.ea_bf, esrc, sc.kin, E);
    // fused K+V projection: one pass over kin
    gemm(s, sc.kin, w.kvT, w.kvb, nullptr, sc.kvf, nullptr, E, 256, 256, 0);
    edge_logits_k<<<cdiv((long)E * 8, 256), 256, 0, s>>>(sc.qf, sc.kvf, edst, sc.logits, E);
    fill(s, sc.m, -3.0e38f, (long)Nd * 8);
    seg_max_k<<<cdiv((long)E * 8, 256), 256, 0, s>>>(sc.logits, edst, sc.m, E);
    fill(s, sc.den, 0.0f, (long)Nd * 8);
    seg_expsum_k<<<cdiv((long)E * 8, 256), 256, 0, s>>>(sc.logits, edst, sc.m, sc.ex, sc.den, E);
    fill(s, sc.agg, 0.0f, (long)Nd * 128);
    seg_scatter_k<<<cdiv((long)E * 128, 256), 256, 0, s>>>(sc.ex, sc.den, sc.kvf, edst, sc.agg, E);
    cvt_f32_bf16<<<cdiv((long)Nd * 128, 256), 256, 0, s>>>(sc.agg, sc.agg_bf, (long)Nd * 128);
    gemm(s, sc.agg_bf, w.o.wT, w.o.b, x /*residual*/, x, nullptr, Nd, 128, 128, 0);
    lnorm(s, x, w.ln_ff, nullptr, sc.xd_bf /*reuse*/, Nd, 0);
    gemm(s, sc.xd_bf, w.f1.wT, w.f1.b, nullptr, nullptr, sc.h1_bf, Nd, 512, 128, 1);
    gemm(s, sc.h1_bf, w.f2.wT, w.f2.b, x /*residual*/, x, nullptr, Nd, 128, 512, 0);
}

// ==================== entry point ====================

extern "C" void kernel_launch(void* const* d_in, const int* in_sizes, int n_in,
                              void* d_out, int out_size, void* d_ws, size_t ws_size,
                              hipStream_t stream) {
    const float* position = (const float*)d_in[0];
    const float* heading  = (const float*)d_in[1];
    const float* velocity = (const float*)d_in[2];
    const unsigned char* vis = (const unsigned char*)d_in[3];
    const float* box      = (const float*)d_in[4];
    const int* atype      = (const int*)d_in[5];
    const int* aid        = (const int*)d_in[6];
    const float* map_emb  = (const float*)d_in[7];
    const float* ppos     = (const float*)d_in[8];
    const float* phead    = (const float*)d_in[9];
    const float* hvalid   = (const float*)d_in[10];
    const int* t_ei = (const int*)d_in[11]; int Et = in_sizes[11] / 2;
    const int* m_ei = (const int*)d_in[12]; int Em = in_sizes[12] / 2;
    const int* a_ei = (const int*)d_in[13]; int Ea = in_sizes[13] / 2;

    // ---- params (jax pytree: dict keys sorted alphabetically) ----
    Cur c{d_in, 14};
    AttnPRaw agent_attn_r[NL_]; for (int i = 0; i < NL_; ++i) agent_attn_r[i] = parse_attn(c);
    MlpP agent_edge_mlp = parse_mlp(c);
    MlpP agent_mlp      = parse_mlp(c);
    const float* id_emb = c.nf();
    AttnPRaw map_attn_r[NL_]; for (int i = 0; i < NL_; ++i) map_attn_r[i] = parse_attn(c);
    MlpP map_edge_mlp   = parse_mlp(c);
    MlpP state_mlp      = parse_mlp(c);
    AttnPRaw temp_attn_r[NL_]; for (int i = 0; i < NL_; ++i) temp_attn_r[i] = parse_attn(c);
    MlpP temporal_edge_mlp = parse_mlp(c);
    const float* type_emb = c.nf();

    Bump ws{(char*)d_ws, 0, ws_size};
    hipStream_t s = stream;

    // ---- weight conversion (fp32 -> bf16, transposed for WMMA B operand) ----
    AttnW aw[NL_], mw[NL_], tw[NL_];
    for (int i = 0; i < NL_; ++i) aw[i] = conv_attn(ws, s, agent_attn_r[i]);
    for (int i = 0; i < NL_; ++i) mw[i] = conv_attn(ws, s, map_attn_r[i]);
    for (int i = 0; i < NL_; ++i) tw[i] = conv_attn(ws, s, temp_attn_r[i]);
    MlpW w_agent_edge = conv_mlp(ws, s, agent_edge_mlp, 5);
    MlpW w_agent      = conv_mlp(ws, s, agent_mlp, 4);
    MlpW w_map_edge   = conv_mlp(ws, s, map_edge_mlp, 6);
    MlpW w_state      = conv_mlp(ws, s, state_mlp, 4);
    MlpW w_temp_edge  = conv_mlp(ws, s, temporal_edge_mlp, 6);

    const int Nn = NA_ * T_;
    long maxE = Et; if (Em > maxE) maxE = Em; if (Ea > maxE) maxE = Ea;

    // ---- workspace buffers ----
    float* emb    = (float*)ws.get((size_t)Nn * 128 * 4);
    float* emb_tm = (float*)ws.get((size_t)Nn * 128 * 4);
    float* stat   = (float*)ws.get((size_t)NA_ * 128 * 4);
    float* t_attr = (float*)ws.get((size_t)Et * 128 * 4);
    float* m_attr = (float*)ws.get((size_t)Em * 128 * 4);
    float* a_attr = (float*)ws.get((size_t)Ea * 128 * 4);
    bf16* node_pad = (bf16*)ws.get((size_t)Nn * 32 * 2);
    bf16* box_pad  = (bf16*)ws.get((size_t)NA_ * 32 * 2);
    bf16* edge_pad = (bf16*)ws.get((size_t)maxE * 32 * 2);

    Scr sc;
    sc.xs_bf  = (bf16*)ws.get((size_t)Nn * 128 * 2);
    sc.xd_bf  = (bf16*)ws.get((size_t)Nn * 128 * 2);
    sc.ea_bf  = (bf16*)ws.get((size_t)maxE * 128 * 2);
    sc.kin    = (bf16*)ws.get((size_t)maxE * 256 * 2);
    sc.agg_bf = (bf16*)ws.get((size_t)Nn * 128 * 2);
    sc.h1_bf  = (bf16*)ws.get((size_t)Nn * 512 * 2);
    sc.qf     = (float*)ws.get((size_t)Nn * 128 * 4);
    sc.kvf    = (float*)ws.get((size_t)maxE * 256 * 4);
    sc.logits = (float*)ws.get((size_t)maxE * 8 * 4);
    sc.ex     = (float*)ws.get((size_t)maxE * 8 * 4);
    sc.m      = (float*)ws.get((size_t)Nn * 8 * 4);
    sc.den    = (float*)ws.get((size_t)Nn * 8 * 4);
    sc.agg    = (float*)ws.get((size_t)Nn * 128 * 4);

    // mlp2 scratch (only used before the attention layers start -> reuse attention scratch)
    float* mscr_f  = sc.kvf;           // >= maxE*128 fp32
    bf16*  mscr_bf = (bf16*)sc.kin;    // >= maxE*128 bf16

    // ---- node embeddings ----
    fill(s, (float*)node_pad, 0.0f, (long)Nn * 16);          // zero bf16 pad (as f32 words)
    state_feat_k<<<cdiv(Nn, 256), 256, 0, s>>>(position, heading, velocity, node_pad);
    run_mlp2(s, w_state, node_pad, emb, Nn, mscr_f, mscr_bf);

    fill(s, (float*)box_pad, 0.0f, (long)NA_ * 16);
    box_pad_k<<<cdiv(NA_ * 4, 256), 256, 0, s>>>(box, box_pad);
    run_mlp2(s, w_agent, box_pad, stat, NA_, mscr_f, mscr_bf);
    add_static_k<<<cdiv((long)Nn * 128, 256), 256, 0, s>>>(emb, stat, type_emb, id_emb, atype, aid);

    // ---- edge attributes ----
    fill(s, (float*)edge_pad, 0.0f, (long)Et * 16);
    temporal_feat_k<<<cdiv(Et, 256), 256, 0, s>>>(t_ei, t_ei + Et, position, heading, edge_pad, Et);
    run_mlp2(s, w_temp_edge, edge_pad, t_attr, Et, mscr_f, mscr_bf);

    fill(s, (float*)edge_pad, 0.0f, (long)Em * 16);
    map_feat_k<<<cdiv(Em, 256), 256, 0, s>>>(m_ei, m_ei + Em, ppos, phead, hvalid,
                                             position, heading, edge_pad, Em);
    run_mlp2(s, w_map_edge, edge_pad, m_attr, Em, mscr_f, mscr_bf);

    fill(s, (float*)edge_pad, 0.0f, (long)Ea * 16);
    agent_feat_k<<<cdiv(Ea, 256), 256, 0, s>>>(a_ei, a_ei + Ea, position, heading, edge_pad, Ea);
    run_mlp2(s, w_agent_edge, edge_pad, a_attr, Ea, mscr_f, mscr_bf);

    // ---- attention layers ----
    for (int l = 0; l < NL_; ++l) {
        run_attention(s, tw[l], emb, Nn, emb, Nn, t_ei, t_ei + Et, Et, t_attr, sc);
        run_attention(s, mw[l], map_emb, NP_, emb, Nn, m_ei, m_ei + Em, Em, m_attr, sc);
        transpose_nodes_k<<<cdiv((long)Nn * 128, 256), 256, 0, s>>>(emb, emb_tm, 1);
        run_attention(s, aw[l], emb_tm, Nn, emb_tm, Nn, a_ei, a_ei + Ea, Ea, a_attr, sc);
        transpose_nodes_k<<<cdiv((long)Nn * 128, 256), 256, 0, s>>>(emb_tm, emb, 0);
    }

    // ---- output: [emb (Na*T*128) | visible_mask (Na*T)] ----
    long n_emb = (long)Nn * 128, n_mask = Nn;
    long total = n_emb + n_mask;
    if (total > out_size) total = out_size;
    write_out_k<<<cdiv(total, 256), 256, 0, s>>>(emb, vis, (float*)d_out, n_emb, n_mask, total);
}
